// TransformerRanker_53953379173075
// MI455X (gfx1250) — matrix operand order repaired and checked by reading, not compiled, not run on hardware
//
#include <hip/hip_runtime.h>
#include <hip/hip_bf16.h>
#include <stdint.h>
#include <stddef.h>

// ============================================================================
// TransformerRanker forward for MI455X (gfx1250), bf16 WMMA path.
//
// Key simplification: seq len is 1 => softmax(scores)==1 => MHA == (x@wv+bv)@wo+bo.
// All GEMMs run as bf16 WMMA (v_wmma_f32_16x16x32_bf16) with fp32 accumulate.
// Weights are repacked on-device each launch into WMMA B-fragment layout.
// ============================================================================

typedef __bf16 bf16_t;
typedef __attribute__((ext_vector_type(16))) __bf16 v16bf;
typedef __attribute__((ext_vector_type(8)))  float  v8f;

#define MROWS 16384
#define LDSW  36          // LDS row stride in bf16 (32 data + 4 pad; 72B -> conflict-free)

// ---------------------------------------------------------------------------
// Pack fp32 row-major W[K,N] into bf16 WMMA B-fragment tiles.
// Tile (kt,nt) covers K[kt*32..+32), N[nt*16..+16). Tile storage: 512 bf16,
// element (lane,j): lane 0-15 -> K=0..15 (j = K offset), lanes 16-31 -> K=16..31.
// Tiles ordered nt-major, kt fastest (streaming in the GEMM k-loop).
// ---------------------------------------------------------------------------
__global__ void pack_w_kernel(const float* __restrict__ src, bf16_t* __restrict__ dst,
                              int K, int N)
{
    int Ktiles = K >> 5;
    size_t total = (size_t)K * N;
    for (size_t idx = (size_t)blockIdx.x * blockDim.x + threadIdx.x; idx < total;
         idx += (size_t)gridDim.x * blockDim.x) {
        size_t tile = idx >> 9;
        int e    = (int)(idx & 511);
        int lane = e >> 4;
        int j    = e & 15;
        int nt = (int)(tile / Ktiles);
        int kt = (int)(tile % Ktiles);
        int k = kt * 32 + (lane >> 4) * 16 + j;
        int n = nt * 16 + (lane & 15);
        dst[idx] = (bf16_t)src[(size_t)k * N + n];
    }
}

// ---------------------------------------------------------------------------
// Embedding gather + concat -> bf16 feats [M, 1088]
// layout per row: user(8*64) | ad(8*64) | numerical(64)
// ---------------------------------------------------------------------------
__global__ void gather_concat_kernel(const int* __restrict__ ucat, const int* __restrict__ acat,
                                     const float* __restrict__ num,
                                     const float* __restrict__ uemb, const float* __restrict__ aemb,
                                     bf16_t* __restrict__ feats)
{
    size_t total = (size_t)MROWS * 1088;
    for (size_t idx = (size_t)blockIdx.x * blockDim.x + threadIdx.x; idx < total;
         idx += (size_t)gridDim.x * blockDim.x) {
        int b = (int)(idx / 1088);
        int c = (int)(idx % 1088);
        float v;
        if (c < 512) {
            int cat = c >> 6, e = c & 63;
            int tok = ucat[b * 8 + cat];
            v = uemb[((size_t)cat * 10000 + tok) * 64 + e];
        } else if (c < 1024) {
            int c2 = c - 512;
            int cat = c2 >> 6, e = c2 & 63;
            int tok = acat[b * 8 + cat];
            v = aemb[((size_t)cat * 10000 + tok) * 64 + e];
        } else {
            v = num[(size_t)b * 64 + (c - 1024)];
        }
        feats[idx] = (bf16_t)v;
    }
}

__global__ void combine_bias_kernel(const float* __restrict__ a, const float* __restrict__ b,
                                    float* __restrict__ out, int n)
{
    int i = blockIdx.x * blockDim.x + threadIdx.x;
    if (i < n) out[i] = a[i] + b[i];
}

// ---------------------------------------------------------------------------
// Main bf16 WMMA GEMM:  C[M,N] = act(A[M,K] @ W + bias (+bias2))
// Block = 256 thr = 8 wave32; tile 128(M) x 64(N); wave -> 32x32 (2x2 frags).
// A staged via LDS; W read from packed fragment layout (coalesced b128).
// M%128==0, N%64==0, K%32==0 (holds for all shapes here).
// ---------------------------------------------------------------------------
__global__ __launch_bounds__(256)
void gemm_bf16_wmma(const bf16_t* __restrict__ A, const bf16_t* __restrict__ Wp,
                    const float* __restrict__ bias, const float* __restrict__ bias2,
                    bf16_t* __restrict__ C, int M, int N, int K, int relu)
{
    __shared__ bf16_t aTile[128 * LDSW];

    const int tid    = threadIdx.x;
    const int lane   = tid & 31;
    const int wave   = tid >> 5;
    const int lrow   = lane & 15;        // fragment row / column index
    const int khalf  = lane >> 4;        // 0 or 1 (half-wave)
    const int mblock = blockIdx.y * 128;
    const int nblock = blockIdx.x * 64;
    const int m_wave = (wave & 3) * 32;
    const int n_wave = (wave >> 2) * 32;
    const int Ktiles = K >> 5;

    // cooperative A-tile load mapping: 256 thr * 16 bf16 = 128x32
    const int ld_row = tid >> 1;
    const int ld_col = (tid & 1) * 16;

    v8f acc[2][2];
    const v8f vzero = {0.f, 0.f, 0.f, 0.f, 0.f, 0.f, 0.f, 0.f};
    acc[0][0] = vzero; acc[0][1] = vzero; acc[1][0] = vzero; acc[1][1] = vzero;

    union AFrag { uint32_t u[8]; v16bf v; };
    union BFrag { uint4 q[2];   v16bf v; };

    for (int kt = 0; kt < Ktiles; ++kt) {
        __syncthreads();
        {   // global (coalesced b128) -> LDS (b64 stores, 8B aligned w/ stride 36)
            const bf16_t* src = A + (size_t)(mblock + ld_row) * K + (size_t)kt * 32 + ld_col;
            uint4 p0 = ((const uint4*)src)[0];
            uint4 p1 = ((const uint4*)src)[1];
            uint2* d2 = (uint2*)&aTile[ld_row * LDSW + ld_col];
            d2[0] = uint2{p0.x, p0.y};
            d2[1] = uint2{p0.z, p0.w};
            d2[2] = uint2{p1.x, p1.y};
            d2[3] = uint2{p1.z, p1.w};
        }
        __syncthreads();

        // A fragments from LDS (contiguous K order: K 16*half..16*half+15 per half-wave)
        AFrag af[2];
#pragma unroll
        for (int fm = 0; fm < 2; ++fm) {
            const bf16_t* p = &aTile[(m_wave + fm * 16 + lrow) * LDSW + khalf * 16];
            uint2 g0a = *(const uint2*)(p);
            uint2 g0b = *(const uint2*)(p + 4);
            uint2 g1a = *(const uint2*)(p + 8);
            uint2 g1b = *(const uint2*)(p + 12);
            af[fm].u[0] = g0a.x; af[fm].u[1] = g0a.y;
            af[fm].u[2] = g0b.x; af[fm].u[3] = g0b.y;
            af[fm].u[4] = g1a.x; af[fm].u[5] = g1a.y;
            af[fm].u[6] = g1b.x; af[fm].u[7] = g1b.y;
        }

        // B fragments: 32 contiguous bytes per lane from packed layout
        BFrag bfg[2];
#pragma unroll
        for (int fn = 0; fn < 2; ++fn) {
            int ntile = ((nblock + n_wave) >> 4) + fn;
            const bf16_t* p = Wp + ((size_t)ntile * Ktiles + kt) * 512 + lane * 16;
            bfg[fn].q[0] = ((const uint4*)p)[0];
            bfg[fn].q[1] = ((const uint4*)p)[1];
        }

#pragma unroll
        for (int fm = 0; fm < 2; ++fm)
#pragma unroll
            for (int fn = 0; fn < 2; ++fn)
                acc[fm][fn] = __builtin_amdgcn_wmma_f32_16x16x32_bf16(
                    false, af[fm].v, false, bfg[fn].v, (short)0, acc[fm][fn], false, false);
    }

    // epilogue: C/D 16x16 f32 layout -> row = r + khalf*8, col = lrow
#pragma unroll
    for (int fm = 0; fm < 2; ++fm) {
        int row0 = mblock + m_wave + fm * 16 + khalf * 8;
#pragma unroll
        for (int fn = 0; fn < 2; ++fn) {
            int col = nblock + n_wave + fn * 16 + lrow;
            float bs = 0.f;
            if (bias)  bs += bias[col];
            if (bias2) bs += bias2[col];
#pragma unroll
            for (int r = 0; r < 8; ++r) {
                float v = acc[fm][fn][r] + bs;
                if (relu) v = fmaxf(v, 0.f);
                C[(size_t)(row0 + r) * N + col] = (bf16_t)v;
            }
        }
    }
}

// ---------------------------------------------------------------------------
// out = LayerNorm(x + h) over 512 dims; one wave32 per row; fp32 math.
// ---------------------------------------------------------------------------
__global__ __launch_bounds__(256)
void residual_ln_kernel(const bf16_t* __restrict__ x, const bf16_t* __restrict__ h,
                        const float* __restrict__ g, const float* __restrict__ b,
                        bf16_t* __restrict__ out)
{
    int lane = threadIdx.x & 31;
    int row  = blockIdx.x * 8 + (threadIdx.x >> 5);
    const bf16_t* xr = x + (size_t)row * 512;
    const bf16_t* hr = h + (size_t)row * 512;
    float vals[16], s = 0.f, s2 = 0.f;
#pragma unroll
    for (int i = 0; i < 16; ++i) {
        int j = lane + i * 32;
        float v = (float)xr[j] + (float)hr[j];
        vals[i] = v; s += v; s2 += v * v;
    }
#pragma unroll
    for (int off = 16; off > 0; off >>= 1) {
        s  += __shfl_xor(s,  off, 32);
        s2 += __shfl_xor(s2, off, 32);
    }
    float mu  = s * (1.f / 512.f);
    float var = s2 * (1.f / 512.f) - mu * mu;
    float inv = rsqrtf(var + 1e-5f);
    bf16_t* orow = out + (size_t)row * 512;
#pragma unroll
    for (int i = 0; i < 16; ++i) {
        int j = lane + i * 32;
        orow[j] = (bf16_t)((vals[i] - mu) * inv * g[j] + b[j]);
    }
}

// xl = x0 * c + xl  (elementwise, fp32 math)
__global__ void cross_update_kernel(const bf16_t* __restrict__ x0, const bf16_t* __restrict__ c,
                                    bf16_t* __restrict__ xl, size_t n)
{
    for (size_t i = (size_t)blockIdx.x * blockDim.x + threadIdx.x; i < n;
         i += (size_t)gridDim.x * blockDim.x)
        xl[i] = (bf16_t)((float)x0[i] * (float)c[i] + (float)xl[i]);
}

// final head: out[row] = dot(y2[row,0:64], w3) + b3
__global__ void head_final_kernel(const bf16_t* __restrict__ y2, const float* __restrict__ w3,
                                  const float* __restrict__ b3, float* __restrict__ out)
{
    int row = blockIdx.x * blockDim.x + threadIdx.x;
    if (row >= MROWS) return;
    const bf16_t* yr = y2 + (size_t)row * 64;
    float acc = b3[0];
#pragma unroll
    for (int j = 0; j < 64; ++j) acc += (float)yr[j] * w3[j];
    out[row] = acc;
}

// ============================================================================
// Host orchestration.
// Input index assumption: top-level dict insertion order
//   [0]=user_categorical(int) [1]=ad_categorical(int) [2]=numerical(f32)
// then params flattened jax-style (sorted dict keys at every level):
//   3=ad_emb, 4=cross_b, 5=cross_w,
//   heads (ctr,engagement,revenue) x (b1,b2,b3,w1,w2,w3) = 6..23,
//   layers[0..2] x (b1,b2,bk,bo,bq,bv,ln1_b,ln1_g,ln2_b,ln2_g,w1,w2,wk,wo,wq,wv) = 24..71,
//   72=pos, 73=proj_b, 74=proj_w, 75=user_emb
// ============================================================================
extern "C" void kernel_launch(void* const* d_in, const int* in_sizes, int n_in,
                              void* d_out, int out_size, void* d_ws, size_t ws_size,
                              hipStream_t stream)
{
    (void)in_sizes; (void)n_in; (void)out_size; (void)ws_size;
    const int M = MROWS;

    const int*   user_cat = (const int*)d_in[0];
    const int*   ad_cat   = (const int*)d_in[1];
    const float* numer    = (const float*)d_in[2];
    const float* ad_emb   = (const float*)d_in[3];
    const float* cross_b  = (const float*)d_in[4];
    const float* cross_w  = (const float*)d_in[5];
    const float* pos      = (const float*)d_in[72];
    const float* proj_b   = (const float*)d_in[73];
    const float* proj_w   = (const float*)d_in[74];
    const float* user_emb = (const float*)d_in[75];

    // workspace layout
    char* ws = (char*)d_ws;
    size_t off = 0;
    auto take = [&](size_t bytes) { size_t o = off; off += (bytes + 255) & ~(size_t)255; return o; };

    // big region shared by feats / h1 / (y1,y2): live ranges are disjoint
    size_t o_big   = take((size_t)M * 2048 * 2);           // 64 MB
    size_t o_feats = o_big;                                 // [M,1088] bf16
    size_t o_h1    = o_big;                                 // [M,2048] bf16
    size_t o_y1    = o_big;                                 // [M,256] bf16
    size_t o_y2    = o_big + (size_t)M * 256 * 2;           // [M,64] bf16
    size_t o_xA = take((size_t)M * 512 * 2);                // running activation
    size_t o_xB = take((size_t)M * 512 * 2);                // temp
    size_t o_xC = take((size_t)M * 512 * 2);                // temp
    size_t o_x0 = take((size_t)M * 512 * 2);                // cross-net x0
    size_t o_cb = take(512 * 4);                            // proj_b + pos
    size_t o_pproj = take((size_t)1088 * 512 * 2);
    size_t o_pv[3], o_po[3], o_p1[3], o_p2[3], o_pc[3], o_pw1[3], o_pw2[3];
    for (int l = 0; l < 3; ++l) {
        o_pv[l] = take((size_t)512 * 512 * 2);
        o_po[l] = take((size_t)512 * 512 * 2);
        o_p1[l] = take((size_t)512 * 2048 * 2);
        o_p2[l] = take((size_t)2048 * 512 * 2);
    }
    for (int i = 0; i < 3; ++i) o_pc[i] = take((size_t)512 * 512 * 2);
    for (int t = 0; t < 3; ++t) { o_pw1[t] = take((size_t)512 * 256 * 2); o_pw2[t] = take((size_t)256 * 64 * 2); }

    auto Wb = [&](size_t o) { return (bf16_t*)(ws + o); };
    auto Ff = [&](size_t o) { return (float*)(ws + o); };

    auto pack = [&](const float* src, size_t dstOff, int K, int N) {
        int blocks = (int)(((size_t)K * N + 255) / 256);
        pack_w_kernel<<<blocks, 256, 0, stream>>>(src, Wb(dstOff), K, N);
    };
    auto gemm = [&](const bf16_t* A, size_t wOff, const float* bias, const float* bias2,
                    bf16_t* C, int N, int K, int relu) {
        dim3 grid(N / 64, M / 128);
        gemm_bf16_wmma<<<grid, 256, 0, stream>>>(A, Wb(wOff), bias, bias2, C, M, N, K, relu);
    };

    // ---- pack all weights (bf16, fragment layout) ----
    pack(proj_w, o_pproj, 1088, 512);
    for (int l = 0; l < 3; ++l) {
        int base = 24 + l * 16;
        pack((const float*)d_in[base + 15], o_pv[l], 512, 512);   // wv
        pack((const float*)d_in[base + 13], o_po[l], 512, 512);   // wo
        pack((const float*)d_in[base + 10], o_p1[l], 512, 2048);  // w1
        pack((const float*)d_in[base + 11], o_p2[l], 2048, 512);  // w2
    }
    for (int i = 0; i < 3; ++i) pack(cross_w + (size_t)i * 512 * 512, o_pc[i], 512, 512);
    for (int t = 0; t < 3; ++t) {
        int hb = 6 + t * 6;
        pack((const float*)d_in[hb + 3], o_pw1[t], 512, 256);     // head w1
        pack((const float*)d_in[hb + 4], o_pw2[t], 256, 64);      // head w2
    }

    // ---- proj bias + positional (seq position 0) ----
    combine_bias_kernel<<<2, 256, 0, stream>>>(proj_b, pos, Ff(o_cb), 512);

    // ---- embedding gather + concat ----
    {
        int blocks = (int)(((size_t)M * 1088 + 255) / 256);
        gather_concat_kernel<<<blocks, 256, 0, stream>>>(user_cat, ad_cat, numer,
                                                         user_emb, ad_emb, Wb(o_feats));
    }

    // ---- input projection: xA = feats @ proj_w + (proj_b + pos) ----
    gemm(Wb(o_feats), o_pproj, Ff(o_cb), nullptr, Wb(o_xA), 512, 1088, 0);

    // ---- 3 encoder layers (seq=1 => attention == v@wo) ----
    for (int l = 0; l < 3; ++l) {
        int base = 24 + l * 16;
        const float* b1    = (const float*)d_in[base + 0];
        const float* b2    = (const float*)d_in[base + 1];
        const float* bo    = (const float*)d_in[base + 3];
        const float* bv    = (const float*)d_in[base + 5];
        const float* ln1_b = (const float*)d_in[base + 6];
        const float* ln1_g = (const float*)d_in[base + 7];
        const float* ln2_b = (const float*)d_in[base + 8];
        const float* ln2_g = (const float*)d_in[base + 9];

        gemm(Wb(o_xA), o_pv[l], bv, nullptr, Wb(o_xB), 512, 512, 0);    // v
        gemm(Wb(o_xB), o_po[l], bo, nullptr, Wb(o_xC), 512, 512, 0);    // attn out
        residual_ln_kernel<<<M / 8, 256, 0, stream>>>(Wb(o_xA), Wb(o_xC), ln1_g, ln1_b, Wb(o_xA));
        gemm(Wb(o_xA), o_p1[l], b1, nullptr, Wb(o_h1), 2048, 512, 1);   // relu(x@w1+b1)
        gemm(Wb(o_h1), o_p2[l], b2, nullptr, Wb(o_xB), 512, 2048, 0);   // @w2+b2
        residual_ln_kernel<<<M / 8, 256, 0, stream>>>(Wb(o_xA), Wb(o_xB), ln2_g, ln2_b, Wb(o_xA));
    }

    // ---- cross network: x0 = x; xl = x0*(xl@W+b) + xl, 3 times ----
    hipMemcpyAsync(Wb(o_x0), Wb(o_xA), (size_t)M * 512 * 2, hipMemcpyDeviceToDevice, stream);
    for (int i = 0; i < 3; ++i) {
        gemm(Wb(o_xA), o_pc[i], cross_b + (size_t)i * 512, nullptr, Wb(o_xB), 512, 512, 0);
        cross_update_kernel<<<8192, 256, 0, stream>>>(Wb(o_x0), Wb(o_xB), Wb(o_xA), (size_t)M * 512);
    }

    // ---- task heads ----
    for (int t = 0; t < 3; ++t) {
        int hb = 6 + t * 6;
        const float* hb1 = (const float*)d_in[hb + 0];
        const float* hb2 = (const float*)d_in[hb + 1];
        const float* hb3 = (const float*)d_in[hb + 2];
        const float* hw3 = (const float*)d_in[hb + 5];
        gemm(Wb(o_xA), o_pw1[t], hb1, nullptr, Wb(o_y1), 256, 512, 1);
        gemm(Wb(o_y1), o_pw2[t], hb2, nullptr, Wb(o_y2), 64, 256, 1);
        head_final_kernel<<<M / 256, 256, 0, stream>>>(Wb(o_y2), hw3, hb3,
                                                       (float*)d_out + (size_t)t * M);
    }
}
